// New3_77395310674432
// MI455X (gfx1250) — matrix operand order repaired and checked by reading, not compiled, not run on hardware
//
#include <hip/hip_runtime.h>
#include <hip/hip_bf16.h>

typedef __attribute__((ext_vector_type(16))) __bf16 v16bf;
typedef __attribute__((ext_vector_type(8)))  float  v8f;

union BF16x16 { v16bf v; uint4 u[2]; };
union U4x8    { uint4 u; unsigned short s[8]; };

#define N_ITEMS 10000
#define NPAD    10112      // 79 * 128
#define D2      256
#define D3      384
#define BATCH   1024
#define HIST    100
#define LPAD    112        // 7 * 16

__device__ __forceinline__ unsigned short f2bf(float x) {
  unsigned u = __float_as_uint(x);
  return (unsigned short)((u + 0x7FFFu + ((u >> 16) & 1u)) >> 16);  // RNE
}
__device__ __forceinline__ float bf2f(unsigned short h) {
  return __uint_as_float(((unsigned)h) << 16);
}
__device__ __forceinline__ v8f vzero() {
  v8f z = {0.f, 0.f, 0.f, 0.f, 0.f, 0.f, 0.f, 0.f};
  return z;
}
__device__ __forceinline__ v8f wmma_bf16(v16bf a, v16bf b, v8f c) {
  return __builtin_amdgcn_wmma_f32_16x16x32_bf16(false, a, false, b, (short)0, c,
                                                 false, false);
}

// A fragment (16x32 bf16) from row-major [rows][pitch] bf16 storage.
// Lane L (<16): row L, K {0..7} and {16..23}; lane L+16: row L, K {8..15},{24..31}.
__device__ __forceinline__ v16bf fragA(const unsigned short* base, int pitch,
                                       int row, int kbase, int lane) {
  int r  = row + (lane & 15);
  int kb = kbase + ((lane >> 4) << 3);
  const unsigned short* p = base + (size_t)r * pitch + kb;
  BF16x16 t;
  t.u[0] = *(const uint4*)(p);
  t.u[1] = *(const uint4*)(p + 16);
  return t.v;
}
// B fragment (32x16 bf16) from N-major [cols][pitch] bf16 storage (contiguous K).
// Lane L (<16): col L, K {0..7} and {16..23}; lane L+16: col L, K {8..15},{24..31}.
__device__ __forceinline__ v16bf fragB(const unsigned short* base, int pitch,
                                       int col, int kbase, int lane) {
  int c  = col + (lane & 15);
  int kb = kbase + ((lane >> 4) << 3);
  const unsigned short* p = base + (size_t)c * pitch + kb;
  BF16x16 t;
  t.u[0] = *(const uint4*)(p);
  t.u[1] = *(const uint4*)(p + 16);
  return t.v;
}

// ---------------- conversion kernels ----------------
// kbf[n][f] = bf16 of k-vector = [emb_out[n], emb_in[n]]; pad rows zero.
// (q-vector of n is kbf[n][(f+128)&255].)
__global__ void cvt_kbf(const float* __restrict__ emb_in,
                        const float* __restrict__ emb_out,
                        unsigned short* __restrict__ kbf) {
  size_t idx = (size_t)blockIdx.x * 256 + threadIdx.x;
  if (idx >= (size_t)NPAD * D2) return;
  int n = (int)(idx >> 8), f = (int)(idx & 255);
  float v = 0.f;
  if (n < N_ITEMS) v = (f < 128) ? emb_out[(size_t)n * 128 + f]
                                 : emb_in[(size_t)n * 128 + (f - 128)];
  kbf[idx] = f2bf(v);
}
__global__ void cvt_item(const float* __restrict__ emb_item,
                         unsigned short* __restrict__ itemb) {
  size_t idx = (size_t)blockIdx.x * 256 + threadIdx.x;
  if (idx >= (size_t)NPAD * 128) return;
  int n = (int)(idx >> 7);
  itemb[idx] = f2bf(n < N_ITEMS ? emb_item[idx] : 0.f);
}
__global__ void cvt_wk(const float* __restrict__ Wk, unsigned short* __restrict__ wkb) {
  int idx = blockIdx.x * 256 + threadIdx.x;
  if (idx >= D3 * D3) return;
  wkb[idx] = f2bf(Wk[idx]);
}

// ---------------- kernel B: streaming (flash) region attention ----------------
// region[n] = sum_key exp(score(n,key)/16) * q[key] / sum_key exp(...)
// 8 waves * 16 query rows = 128 rows/block. Keys streamed in tiles of 32.
// LDS byte layout (dynamic LDS, no static -> base offset 0):
//   sQ  [128][256] bf16 : [0, 65536)
//   sK  [ 32][256] bf16 : [65536, 81920)   <- filled by async DMA loads
//   sVT [256][ 32] bf16 : [81920, 98304)
//   sP  [8][16][32] bf16: [98304, 106496)
#define SK_BYTE_OFF 65536u

__global__ void __launch_bounds__(256)
region_attn_kernel(const unsigned short* __restrict__ kbf,
                   float* __restrict__ region_f32,
                   unsigned short* __restrict__ region_bf16) {
  extern __shared__ char smem[];
  unsigned short* sQ  = (unsigned short*)smem;        // [128][256] q-vectors
  unsigned short* sK  = sQ + 128 * 256;               // [32][256]  k-vectors (B of QK^T)
  unsigned short* sVT = sK + 32 * 256;                // [256][32]  q-vectors^T (B of PV)
  unsigned short* sP  = sVT + 256 * 32;               // [8 waves][16][32] exp(scores)

  const int tid = threadIdx.x, lane = tid & 31, wave = tid >> 5;
  const int qblock = blockIdx.x * 128;

  // Stage this block's 128 query rows (q = kbf with halves rotated by 128).
  for (int i = tid; i < 128 * 32; i += 256) {
    int r = i >> 5, f0 = (i & 31) * 8;
    int fs = (f0 + 128) & 255;
    uint4 d = *(const uint4*)(kbf + (size_t)(qblock + r) * D2 + fs);
    *(uint4*)(sQ + r * D2 + f0) = d;
  }
  __syncthreads();

  unsigned short* sQw = sQ + wave * 16 * D2;
  unsigned short* sPw = sP + wave * 16 * 32;

  v8f acc[16];
#pragma unroll
  for (int i = 0; i < 16; ++i) acc[i] = vzero();
  float dsum[8] = {0.f, 0.f, 0.f, 0.f, 0.f, 0.f, 0.f, 0.f};

  const float sc = 0.0625f;  // 1/sqrt(2D) = 1/16
  for (int kt = 0; kt < NPAD / 32; ++kt) {
    __syncthreads();
    const int kb0 = kt * 32;
    // (1) Row-major key tile -> LDS via async DMA (no VGPR round-trip,
    //     tracked on ASYNCcnt, overlaps with the transpose staging below).
    for (int i = tid; i < 32 * 32; i += 256) {
      int kk = i >> 5, f0 = (i & 31) * 8;
      unsigned lds_off = SK_BYTE_OFF + (unsigned)(kk * D2 + f0) * 2u;
      unsigned long long gaddr =
          (unsigned long long)(const void*)(kbf + (size_t)(kb0 + kk) * D2 + f0);
      asm volatile("global_load_async_to_lds_b128 %0, %1, off"
                   :: "v"(lds_off), "v"(gaddr)
                   : "memory");
    }
    // (2) Rotated transpose tile (V == q-vectors) via regular loads.
    for (int i = tid; i < 32 * 32; i += 256) {
      int kk = i >> 5, f0 = (i & 31) * 8;
      U4x8 d;
      d.u = *(const uint4*)(kbf + (size_t)(kb0 + kk) * D2 + f0);
      int fswap = (f0 + 128) & 255;
#pragma unroll
      for (int j = 0; j < 8; ++j) sVT[(fswap + j) * 32 + kk] = d.s[j];
    }
    if (lane == 0 && kt + 1 < NPAD / 32)
      __builtin_prefetch(kbf + (size_t)(kb0 + 32) * D2, 0, 1);
    asm volatile("s_wait_asynccnt 0x0" ::: "memory");
    __syncthreads();

    // scores: two 16x16 tiles over K=256 (8 bf16 WMMA steps each)
    v8f s0 = vzero(), s1 = vzero();
#pragma unroll
    for (int fs = 0; fs < 8; ++fs) {
      v16bf a  = fragA(sQw, D2, 0, fs * 32, lane);
      v16bf b0 = fragB(sK, D2, 0, fs * 32, lane);
      v16bf b1 = fragB(sK, D2, 16, fs * 32, lane);
      s0 = wmma_bf16(a, b0, s0);
      s1 = wmma_bf16(a, b1, s1);
    }
    // exp + tail mask + denominator partials + stage P (16x32 bf16)
    const int row_off = (lane >> 4) << 3;
    const int col = lane & 15;
    const int key0 = kb0 + col, key1 = kb0 + 16 + col;
#pragma unroll
    for (int v = 0; v < 8; ++v) {
      float e0 = __expf(s0[v] * sc);
      if (key0 >= N_ITEMS) e0 = 0.f;
      float e1 = __expf(s1[v] * sc);
      if (key1 >= N_ITEMS) e1 = 0.f;
      dsum[v] += e0 + e1;
      int r = v + row_off;
      sPw[r * 32 + col] = f2bf(e0);
      sPw[r * 32 + 16 + col] = f2bf(e1);
    }
    // numerator += P (16x32) @ V (32x256), 16 N-tiles
    v16bf pa = fragA(sPw, 32, 0, 0, lane);
#pragma unroll
    for (int nt = 0; nt < 16; ++nt) {
      v16bf bm = fragB(sVT, 32, nt * 16, 0, lane);
      acc[nt] = wmma_bf16(pa, bm, acc[nt]);
    }
  }

  // Row denominators: reduce across each 16-lane half; lanes then hold exactly
  // the denominator of the rows their C-layout elements belong to.
#pragma unroll
  for (int v = 0; v < 8; ++v) {
    float s = dsum[v];
#pragma unroll
    for (int off = 8; off > 0; off >>= 1) s += __shfl_xor(s, off, 16);
    dsum[v] = s;
  }
  const int row_off = (lane >> 4) << 3;
  const int col = lane & 15;
#pragma unroll
  for (int nt = 0; nt < 16; ++nt) {
#pragma unroll
    for (int v = 0; v < 8; ++v) {
      int qr = qblock + wave * 16 + v + row_off;
      if (qr < N_ITEMS) {
        float val = acc[nt][v] / dsum[v];
        int c = nt * 16 + col;
        region_f32[(size_t)qr * D2 + c] = val;
        region_bf16[(size_t)qr * D2 + c] = f2bf(val);
      }
    }
  }
}

// ---------------- kernel C: fused attention network, one block per batch elem --
__global__ void __launch_bounds__(256)
attn_net_kernel(const int* __restrict__ user,
                const int* __restrict__ item_i,
                const int* __restrict__ item_j,
                const float* __restrict__ emb_item,
                const float* __restrict__ Wq,
                const float* __restrict__ bq,
                const float* __restrict__ bk,
                const float* __restrict__ Wv,
                const float* __restrict__ bv,
                const unsigned short* __restrict__ wk_bf16,
                const unsigned short* __restrict__ item_bf16,
                const float* __restrict__ region_f32,
                const unsigned short* __restrict__ region_bf16,
                float* __restrict__ out) {
  extern __shared__ char smem[];
  unsigned short* sH = (unsigned short*)smem;  // [112][384] bf16 history
  __shared__ float tpos[D3], tneg[D3], qpos[D3], qneg[D3], gpos[D3], gneg[D3];
  __shared__ float sBk[D3], dVp[128], dVn[128], sSp[128], sSn[128], sBvt[2];

  const int b = blockIdx.x;
  const int tid = threadIdx.x, lane = tid & 31, wave = tid >> 5;
  const int ii = item_i[b], jj = item_j[b];

  if (tid < 2) sBvt[tid] = 0.f;
  if (tid < 128) { sSp[tid] = 0.f; sSn[tid] = 0.f; dVp[tid] = 0.f; dVn[tid] = 0.f; }
  // targets (f32) and bk
  for (int d = tid; d < D3; d += 256) {
    tpos[d] = (d < 128) ? emb_item[(size_t)ii * 128 + d]
                        : region_f32[(size_t)ii * D2 + (d - 128)];
    tneg[d] = (d < 128) ? emb_item[(size_t)jj * 128 + d]
                        : region_f32[(size_t)jj * D2 + (d - 128)];
    sBk[d] = bk[d];
  }
  // gather history rows (bf16), zero-pad rows 100..111
  for (int i = tid; i < LPAD * 48; i += 256) {
    int r = i / 48, f0 = (i % 48) * 8;
    uint4 d = make_uint4(0u, 0u, 0u, 0u);
    if (r < HIST) {
      int u = user[b * HIST + r];
      d = (f0 < 128) ? *(const uint4*)(item_bf16 + (size_t)u * 128 + f0)
                     : *(const uint4*)(region_bf16 + (size_t)u * D2 + (f0 - 128));
    }
    *(uint4*)(sH + r * D3 + f0) = d;
  }
  __syncthreads();

  // q = Wq t + bq (rows of Wq), g = Wv^T t (cols of Wv); val_emb never materialized.
  for (int d = tid; d < D3; d += 256) {
    float ap = 0.f, an = 0.f;
    const float* wr = Wq + (size_t)d * D3;
    for (int k = 0; k < D3; ++k) { float w = wr[k]; ap += w * tpos[k]; an += w * tneg[k]; }
    qpos[d] = ap + bq[d];
    qneg[d] = an + bq[d];
    float gp = 0.f, gn = 0.f;
    for (int r = 0; r < D3; ++r) { float w = Wv[(size_t)r * D3 + d]; gp += w * tpos[r]; gn += w * tneg[r]; }
    gpos[d] = gp;
    gneg[d] = gn;
  }
  {
    float pp = 0.f, pn = 0.f;
    for (int d = tid; d < D3; d += 256) { pp += bv[d] * tpos[d]; pn += bv[d] * tneg[d]; }
    atomicAdd(&sBvt[0], pp);
    atomicAdd(&sBvt[1], pn);
  }
  __syncthreads();

  // dotV[l] = H[l] . g + bv.t
  for (int t = tid; t < 2 * HIST; t += 256) {
    int l = t % HIST, br = t / HIST;
    const float* g = br ? gneg : gpos;
    const unsigned short* hr = sH + (size_t)l * D3;
    float s = 0.f;
    for (int k = 0; k < D3; ++k) s += bf2f(hr[k]) * g[k];
    if (br) dVn[l] = s + sBvt[1]; else dVp[l] = s + sBvt[0];
  }
  __syncthreads();

  // K = H @ Wk^T + bk via bf16 WMMA (7x24 tiles, K=384 in 12 steps).
  // Reshape quirk: flat idx i = l'*384+d' feeds s[i%100] with weight q[i/100].
  for (int tl = wave; tl < 7 * 24; tl += 8) {
    int mt = tl / 24, nt = tl % 24;
    v8f acc = vzero();
#pragma unroll
    for (int ks = 0; ks < 12; ++ks) {
      v16bf a  = fragA(sH, D3, mt * 16, ks * 32, lane);
      v16bf bm = fragB(wk_bf16, D3, nt * 16, ks * 32, lane);
      acc = wmma_bf16(a, bm, acc);
    }
    const int row_off = (lane >> 4) << 3;
    const int dcol = nt * 16 + (lane & 15);
#pragma unroll
    for (int v = 0; v < 8; ++v) {
      int lrow = mt * 16 + v + row_off;
      if (lrow < HIST) {
        float kv = acc[v] + sBk[dcol];
        int fi = lrow * D3 + dcol;
        int dd = fi / 100;
        int ll = fi - dd * 100;
        atomicAdd(&sSp[ll], qpos[dd] * kv);
        atomicAdd(&sSn[ll], qneg[dd] * kv);
      }
    }
  }
  __syncthreads();

  if (tid < 32) {
    const float rsc = 0.05103103630798287f;  // 1/sqrt(384)
    float sA = 0.f, sAV = 0.f, sB = 0.f, sBV = 0.f;
    for (int l = lane; l < HIST; l += 32) {
      float ea = __expf(sSp[l] * rsc);
      if (user[b * HIST + l] == ii) ea = 0.f;  // exclusion mask (pos branch only)
      float eb = __expf(sSn[l] * rsc);
      sA += ea; sAV += ea * dVp[l];
      sB += eb; sBV += eb * dVn[l];
    }
#pragma unroll
    for (int off = 16; off > 0; off >>= 1) {
      sA += __shfl_xor(sA, off, 32);
      sAV += __shfl_xor(sAV, off, 32);
      sB += __shfl_xor(sB, off, 32);
      sBV += __shfl_xor(sBV, off, 32);
    }
    if (lane == 0) {
      out[b] = sAV / sqrtf(sA);           // pow(sum, 0.5) with BETA=0.5
      out[BATCH + b] = sBV / sqrtf(sB);
    }
  }
}

extern "C" void kernel_launch(void* const* d_in, const int* in_sizes, int n_in,
                              void* d_out, int out_size, void* d_ws, size_t ws_size,
                              hipStream_t stream) {
  (void)in_sizes; (void)n_in; (void)out_size; (void)ws_size;
  const int* user     = (const int*)d_in[0];
  const int* item_i   = (const int*)d_in[1];
  const int* item_j   = (const int*)d_in[2];
  const float* emb_item = (const float*)d_in[3];
  const float* emb_in   = (const float*)d_in[4];
  const float* emb_out  = (const float*)d_in[5];
  const float* Wq = (const float*)d_in[6];
  const float* bq = (const float*)d_in[7];
  const float* Wk = (const float*)d_in[8];
  const float* bk = (const float*)d_in[9];
  const float* Wv = (const float*)d_in[10];
  const float* bv = (const float*)d_in[11];
  float* out = (float*)d_out;

  char* ws = (char*)d_ws;
  size_t off = 0;
  unsigned short* kbf   = (unsigned short*)(ws + off); off += (size_t)NPAD * D2 * 2;
  unsigned short* itemb = (unsigned short*)(ws + off); off += (size_t)NPAD * 128 * 2;
  unsigned short* wkb   = (unsigned short*)(ws + off); off += (size_t)D3 * D3 * 2;
  float* region         = (float*)(ws + off);          off += (size_t)NPAD * D2 * 4;
  unsigned short* regionb = (unsigned short*)(ws + off); off += (size_t)NPAD * D2 * 2;

  { size_t n = (size_t)NPAD * D2;  cvt_kbf<<<(unsigned)((n + 255) / 256), 256, 0, stream>>>(emb_in, emb_out, kbf); }
  { size_t n = (size_t)NPAD * 128; cvt_item<<<(unsigned)((n + 255) / 256), 256, 0, stream>>>(emb_item, itemb); }
  { int n = D3 * D3;               cvt_wk<<<(n + 255) / 256, 256, 0, stream>>>(Wk, wkb); }

  size_t smemB = (size_t)(128 * 256 + 32 * 256 + 256 * 32 + 8 * 16 * 32) * 2;
  region_attn_kernel<<<NPAD / 128, 256, smemB, stream>>>(kbf, region, regionb);

  size_t smemC = (size_t)LPAD * D3 * 2;
  attn_net_kernel<<<BATCH, 256, smemC, stream>>>(user, item_i, item_j, emb_item,
      Wq, bq, bk, Wv, bv, wkb, itemb, region, regionb, out);
}
